// EquivariantAttention_75436805587446
// MI455X (gfx1250) — matrix operand order repaired and checked by reading, not compiled, not run on hardware
//
#include <hip/hip_runtime.h>
#include <cstdint>

// ---------------------------------------------------------------------------
// EquivariantAttention for MI455X (gfx1250):
//  - all GEMM stages on v_wmma_f32_16x16x32_bf16 (wave32)
//  - global->LDS staging via GLOBAL_LOAD_ASYNC_TO_LDS_B128, double-buffered
//    (ASYNCcnt pipelining: s_wait_asynccnt 8 keeps next tile in flight)
//  - f32 online softmax, flash-attention streaming over keys
// H=8, DH=64, B=2, T=2048, D=512, MAX_REL=32, SCALE=0.125
// ---------------------------------------------------------------------------

typedef __bf16 bhalf;
typedef __attribute__((ext_vector_type(16))) __bf16 v16bf;
typedef __attribute__((ext_vector_type(8)))  __bf16 v8bf;
typedef __attribute__((ext_vector_type(4)))  __bf16 v4bf;
typedef __attribute__((ext_vector_type(8)))  float  v8f;

#define T_SEQ 2048
#define D_MODEL 512
#define N_ROWS 4096   // B*T
#define N_HEADS 8
#define D_HEAD 64

__device__ __forceinline__ v8f wmma_bf16(v16bf a, v16bf b, v8f c) {
  // D = A(16x32 bf16) * B(32x16 bf16) + C(16x16 f32)
  return __builtin_amdgcn_wmma_f32_16x16x32_bf16(
      /*neg_a=*/false, a, /*neg_b=*/false, b,
      /*c_mod=*/(short)0, c, /*reuse_a=*/false, /*reuse_b=*/false);
}

// A-fragment: per lane the 16 bf16 live as [k .. k+7] and [k+8 .. k+15]
__device__ __forceinline__ v16bf ld16_split(const bhalf* p0, const bhalf* p1) {
  v8bf lo = *(const v8bf*)p0;
  v8bf hi = *(const v8bf*)p1;
  v16bf r;
#pragma unroll
  for (int i = 0; i < 8; i++) { r[i] = lo[i]; r[8 + i] = hi[i]; }
  return r;
}

// Async DMA: 16 bytes per lane, global -> LDS, tracked by ASYNCcnt.
// LDS operand = wave-relative byte offset = low 32 bits of the flat address.
__device__ __forceinline__ void async_b128(void* lds, const void* gptr) {
  asm volatile("global_load_async_to_lds_b128 %0, %1, off"
               :: "v"((uint32_t)(uintptr_t)lds),
                  "v"((uint64_t)(uintptr_t)gptr)
               : "memory");
}
__device__ __forceinline__ void wait_async_le8() {
  asm volatile("s_wait_asynccnt 0x8" ::: "memory");
}
__device__ __forceinline__ void wait_async_0() {
  asm volatile("s_wait_asynccnt 0x0" ::: "memory");
}

// ---------------------------------------------------------------------------
// f32 -> bf16 bulk convert (one float4 -> one v4bf per thread)
// ---------------------------------------------------------------------------
__global__ __launch_bounds__(256) void cvt_bf16_kernel(
    const float* __restrict__ in, bhalf* __restrict__ out) {
  const int i = blockIdx.x * 256 + threadIdx.x;
  const float4 v = ((const float4*)in)[i];
  v4bf b;
  b[0] = (bhalf)v.x; b[1] = (bhalf)v.y; b[2] = (bhalf)v.z; b[3] = (bhalf)v.w;
  ((v4bf*)out)[i] = b;
}

// ---------------------------------------------------------------------------
// f32 W(k,c) -> bf16 Wt(c,k): transpose + convert (weights are tiny, 1 MB)
// ---------------------------------------------------------------------------
__global__ __launch_bounds__(256) void cvt_wt_kernel(
    const float* __restrict__ W, bhalf* __restrict__ Wt) {
  const int idx = blockIdx.x * 256 + threadIdx.x;  // k*512 + c
  const int k = idx >> 9, c = idx & 511;
  Wt[(size_t)c * D_MODEL + k] = (bhalf)W[idx];
}

// ---------------------------------------------------------------------------
// Shared GEMM: X(bf16 row-major 4096x512) * Wt(bf16, [c][k]) + bias.
// Block = 128 threads (4 waves); tile 64x64; K in 64-wide slabs,
// double-buffered through LDS with async global->LDS copies.
// mode 0: bf16 row-major out | mode 1: bf16 (b,h,d,t) out | mode 2: f32 out
// ---------------------------------------------------------------------------
__global__ __launch_bounds__(128) void gemm_bf16(
    const bhalf* __restrict__ X, const bhalf* __restrict__ Wt,
    const float* __restrict__ bias, bhalf* __restrict__ outb,
    float* __restrict__ outf, int mode) {
  __shared__ __align__(128) bhalf lx[2][64 * 64];  // [row][k]
  __shared__ __align__(128) bhalf lw[2][64 * 64];  // [col][k]
  const int tid = threadIdx.x;
  const int w  = tid >> 5;
  const int L  = tid & 31;
  const int hl = L >> 4;
  const int ln = L & 15;
  const int row0 = blockIdx.x * 64;
  const int c0   = blockIdx.y * 64;

  auto stage = [&](int buf, int k0) {  // 8 async ops per wave
#pragma unroll
    for (int p = 0; p < 4; p++) {
      const int idx = (tid + p * 128) * 8;  // == r*64 + kk
      const int r = idx >> 6, kk = idx & 63;
      async_b128(&lx[buf][idx], &X[(size_t)(row0 + r) * D_MODEL + k0 + kk]);
      async_b128(&lw[buf][idx], &Wt[(size_t)(c0 + r) * D_MODEL + k0 + kk]);
    }
  };

  v8f acc[4] = {};
  stage(0, 0);

  for (int it = 0; it < D_MODEL / 64; it++) {
    if (it + 1 < D_MODEL / 64) {
      stage((it + 1) & 1, (it + 1) * 64);
      wait_async_le8();   // current slab done; next slab stays in flight
    } else {
      wait_async_0();
    }
    __syncthreads();

    const bhalf* bx = lx[it & 1];
    const bhalf* bw = lw[it & 1];
    const int arow  = w * 16 + ln;
    const int kbase = hl * 16;
#pragma unroll
    for (int kc = 0; kc < 2; kc++) {
      v16bf a = ld16_split(&bx[arow * 64 + kc * 32 + kbase],
                           &bx[arow * 64 + kc * 32 + kbase + 8]);
#pragma unroll
      for (int cT = 0; cT < 4; cT++) {
        const int cc = cT * 16 + ln;
        v16bf bb = *(const v16bf*)&bw[cc * 64 + kc * 32 + hl * 16];
        acc[cT] = wmma_bf16(a, bb, acc[cT]);
      }
    }
    __syncthreads();  // all waves done reading before buffer is re-staged
  }

#pragma unroll
  for (int cT = 0; cT < 4; cT++) {
    const int c = c0 + cT * 16 + ln;
    const float bv = bias[c];
#pragma unroll
    for (int i = 0; i < 8; i++) {
      const int n = row0 + w * 16 + i + 8 * hl;  // global row (b*T + t)
      const float val = acc[cT][i] + bv;
      if (mode == 0) {
        outb[(size_t)n * D_MODEL + c] = (bhalf)val;
      } else if (mode == 1) {
        const int b = n >> 11, t = n & (T_SEQ - 1);
        const int hh = c >> 6, d = c & 63;
        outb[(((size_t)b * N_HEADS + hh) * D_HEAD + d) * T_SEQ + t] = (bhalf)val;
      } else {
        outf[(size_t)n * D_MODEL + c] = val;
      }
    }
  }
}

// ---------------------------------------------------------------------------
// Flash attention. Block = 4 waves sharing one (b,h); each wave owns 16 query
// rows; K/V key-chunks (64 wide) async-staged into block LDS, double-buffered.
// Q/K row-major bf16, V transposed (b,h,d,t) bf16, rel_bias f32.
// ---------------------------------------------------------------------------
__global__ __launch_bounds__(128) void attn_kernel(
    const bhalf* __restrict__ Qb, const bhalf* __restrict__ Kb,
    const bhalf* __restrict__ Vt, const float* __restrict__ rel_bias,
    bhalf* __restrict__ Ob) {
  __shared__ __align__(128) bhalf lk[2][64 * 64];     // [tk][d]
  __shared__ __align__(128) bhalf lv[2][64 * 64];     // [d][tk]
  __shared__ __align__(128) bhalf plds[4][16 * 64];   // per-wave P tile
  __shared__ float relb[65];

  const int tid = threadIdx.x;
  const int w  = tid >> 5;
  const int L  = tid & 31;
  const int hl = L >> 4;
  const int ln = L & 15;
  const int b = blockIdx.z, h = blockIdx.y;
  const int q0 = blockIdx.x * 64 + w * 16;

  const size_t nbase = (size_t)b * T_SEQ;
  const bhalf* vbase = Vt + ((size_t)b * N_HEADS + h) * D_HEAD * (size_t)T_SEQ;

  auto stageKV = [&](int buf, int k0) {  // 8 async ops per wave
#pragma unroll
    for (int p = 0; p < 4; p++) {
      const int idx = (tid + p * 128) * 8;
      const int r = idx >> 6, cc = idx & 63;
      async_b128(&lk[buf][idx],
                 &Kb[(nbase + k0 + r) * D_MODEL + h * D_HEAD + cc]);
      async_b128(&lv[buf][idx], &vbase[(size_t)r * T_SEQ + k0 + cc]);
    }
  };

  if (tid < 65) relb[tid] = rel_bias[h * 65 + tid];
  stageKV(0, 0);
  __syncthreads();
  const float bias_lo = relb[0];   // rel <= -MAX_REL
  const float bias_hi = relb[64];  // rel >=  MAX_REL

  // Q A-fragments (persist across key loop): row = ln, kbase = hl*16
  v16bf aQ[2];
  {
    const bhalf* qrow = Qb + (nbase + q0 + ln) * D_MODEL + h * D_HEAD;
    const int kbase = hl * 16;
#pragma unroll
    for (int dc = 0; dc < 2; dc++)
      aQ[dc] = ld16_split(qrow + dc * 32 + kbase, qrow + dc * 32 + kbase + 8);
  }

  v8f O[4] = {};
  float m[8], l[8];
#pragma unroll
  for (int i = 0; i < 8; i++) { m[i] = -3.0e38f; l[i] = 0.0f; }

  bhalf* pw = plds[w];
  const int NC = T_SEQ / 64;

  for (int it = 0; it < NC; it++) {
    const int k0 = it * 64;
    if (it + 1 < NC) {
      stageKV((it + 1) & 1, k0 + 64);
      wait_async_le8();
    } else {
      wait_async_0();
    }
    __syncthreads();
    const bhalf* ck = lk[it & 1];
    const bhalf* cv = lv[it & 1];

    // --- S = Q * K^T : 4 accumulators of 16x16 over contraction DH=64 ---
    v8f S[4];
#pragma unroll
    for (int t4 = 0; t4 < 4; t4++) {
      v8f s = {};
#pragma unroll
      for (int dc = 0; dc < 2; dc++) {
        v16bf bk = *(const v16bf*)&ck[(t4 * 16 + ln) * 64 + dc * 32 + hl * 16];
        s = wmma_bf16(aQ[dc], bk, s);
      }
      S[t4] = s;
    }

    // --- scale + clamped relative-position bias ---
    // Fully-clamped chunks (all but ~3 near the diagonal) use a wave-uniform
    // scalar bias; only near-diagonal chunks pay the per-element path.
    const int dmin = q0 - (k0 + 63);
    const int dmax = (q0 + 15) - k0;
    if (dmin >= 32) {
#pragma unroll
      for (int t4 = 0; t4 < 4; t4++)
#pragma unroll
        for (int i = 0; i < 8; i++) S[t4][i] = S[t4][i] * 0.125f + bias_hi;
    } else if (dmax <= -32) {
#pragma unroll
      for (int t4 = 0; t4 < 4; t4++)
#pragma unroll
        for (int i = 0; i < 8; i++) S[t4][i] = S[t4][i] * 0.125f + bias_lo;
    } else {
#pragma unroll
      for (int t4 = 0; t4 < 4; t4++) {
        const int tk = k0 + t4 * 16 + ln;
#pragma unroll
        for (int i = 0; i < 8; i++) {
          int rel = (q0 + i + 8 * hl) - tk;
          rel = rel < -32 ? -32 : (rel > 32 ? 32 : rel);
          S[t4][i] = S[t4][i] * 0.125f + relb[rel + 32];
        }
      }
    }

    // --- online softmax: rows are striped over 16-lane halves ---
    float cmax[8];
#pragma unroll
    for (int i = 0; i < 8; i++) {
      float v = fmaxf(fmaxf(S[0][i], S[1][i]), fmaxf(S[2][i], S[3][i]));
#pragma unroll
      for (int off = 1; off < 16; off <<= 1) v = fmaxf(v, __shfl_xor(v, off, 32));
      cmax[i] = v;
    }
#pragma unroll
    for (int i = 0; i < 8; i++) {
      const float mn = fmaxf(m[i], cmax[i]);
      const float corr = __expf(m[i] - mn);
      m[i] = mn;
      float psum = 0.0f;
#pragma unroll
      for (int t4 = 0; t4 < 4; t4++) {
        const float p = __expf(S[t4][i] - mn);
        S[t4][i] = p;
        psum += p;
      }
#pragma unroll
      for (int off = 1; off < 16; off <<= 1) psum += __shfl_xor(psum, off, 32);
      l[i] = l[i] * corr + psum;
#pragma unroll
      for (int nt = 0; nt < 4; nt++) O[nt][i] *= corr;
    }

    // --- C-layout P -> A-fragment layout via per-wave LDS tile ---
#pragma unroll
    for (int t4 = 0; t4 < 4; t4++)
#pragma unroll
      for (int i = 0; i < 8; i++)
        pw[(i + 8 * hl) * 64 + t4 * 16 + ln] = (bhalf)S[t4][i];
    asm volatile("s_wait_dscnt 0x0" ::: "memory");  // cross-lane LDS RAW

    v16bf aP[2];
    {
      const int kbase = hl * 16;
#pragma unroll
      for (int kc = 0; kc < 2; kc++)
        aP[kc] = ld16_split(&pw[ln * 64 + kc * 32 + kbase],
                            &pw[ln * 64 + kc * 32 + kbase + 8]);
    }

    // --- O += P * V (V chunk staged [d][tk]: contiguous along key axis) ---
#pragma unroll
    for (int nt = 0; nt < 4; nt++) {
#pragma unroll
      for (int kc = 0; kc < 2; kc++) {
        v16bf bv = *(const v16bf*)&cv[(nt * 16 + ln) * 64 + kc * 32 + hl * 16];
        O[nt] = wmma_bf16(aP[kc], bv, O[nt]);
      }
    }
    __syncthreads();  // all waves done reading before buffer is re-staged
  }

  // --- normalize and emit bf16 attention output, row-major (B*T, 512) ---
#pragma unroll
  for (int i = 0; i < 8; i++) {
    const float inv = 1.0f / l[i];
    const int n = q0 + i + 8 * hl;
#pragma unroll
    for (int nt = 0; nt < 4; nt++)
      Ob[(nbase + n) * D_MODEL + h * D_HEAD + nt * 16 + ln] =
          (bhalf)(O[nt][i] * inv);
  }
}

// ---------------------------------------------------------------------------
extern "C" void kernel_launch(void* const* d_in, const int* in_sizes, int n_in,
                              void* d_out, int out_size, void* d_ws,
                              size_t ws_size, hipStream_t stream) {
  const float* X   = (const float*)d_in[0];
  const float* Wq  = (const float*)d_in[1];
  const float* bq  = (const float*)d_in[2];
  const float* Wk  = (const float*)d_in[3];
  const float* bk  = (const float*)d_in[4];
  const float* Wv  = (const float*)d_in[5];
  const float* bv  = (const float*)d_in[6];
  const float* Wo  = (const float*)d_in[7];
  const float* bo  = (const float*)d_in[8];
  const float* rb  = (const float*)d_in[9];
  float* out = (float*)d_out;

  const size_t mat  = (size_t)N_ROWS * D_MODEL;    // 2M bf16 elems = 4 MB
  const size_t wmat = (size_t)D_MODEL * D_MODEL;   // 256K bf16 elems
  bhalf* Xb  = (bhalf*)d_ws;
  bhalf* Qb  = Xb + mat;
  bhalf* Kb  = Qb + mat;
  bhalf* Vt  = Kb + mat;   // stored (b, h, d, t)
  bhalf* Ob  = Vt + mat;
  bhalf* Wqt = Ob + mat;
  bhalf* Wkt = Wqt + wmat;
  bhalf* Wvt = Wkt + wmat;
  bhalf* Wot = Wvt + wmat;

  // precision/layout pre-passes (tiny)
  cvt_bf16_kernel<<<(N_ROWS * D_MODEL / 4) / 256, 256, 0, stream>>>(X, Xb);
  cvt_wt_kernel<<<(D_MODEL * D_MODEL) / 256, 256, 0, stream>>>(Wq, Wqt);
  cvt_wt_kernel<<<(D_MODEL * D_MODEL) / 256, 256, 0, stream>>>(Wk, Wkt);
  cvt_wt_kernel<<<(D_MODEL * D_MODEL) / 256, 256, 0, stream>>>(Wv, Wvt);
  cvt_wt_kernel<<<(D_MODEL * D_MODEL) / 256, 256, 0, stream>>>(Wo, Wot);

  dim3 blk(128);
  dim3 gGemm(N_ROWS / 64, D_MODEL / 64, 1);
  gemm_bf16<<<gGemm, blk, 0, stream>>>(Xb, Wqt, bq, Qb, nullptr, 0);
  gemm_bf16<<<gGemm, blk, 0, stream>>>(Xb, Wkt, bk, Kb, nullptr, 0);
  gemm_bf16<<<gGemm, blk, 0, stream>>>(Xb, Wvt, bv, Vt, nullptr, 1);

  dim3 gAttn(T_SEQ / 64, N_HEADS, 2);
  attn_kernel<<<gAttn, blk, 0, stream>>>(Qb, Kb, Vt, rb, Ob);

  gemm_bf16<<<gGemm, blk, 0, stream>>>(Ob, Wot, bo, nullptr, out, 2);
}
